// CasualSelfAttention_36352603193619
// MI455X (gfx1250) — compile-verified, matrix operand order checked by reading, not run on hardware
//
#include <hip/hip_runtime.h>

typedef __bf16 bf16;
typedef __attribute__((ext_vector_type(16))) __bf16 v16bf;
typedef __attribute__((ext_vector_type(8)))  __bf16 v8bf;
typedef __attribute__((ext_vector_type(4)))  __bf16 v4bf;
typedef __attribute__((ext_vector_type(8)))  float  v8f;

#define B_ 8
#define T_ 1024
#define C_ 1024
#define H_ 16
#define D_ 64
#define M_ (B_*T_)

// ---------------------------------------------------------------------------
// WMMA helper: D = A(16x32 bf16) * B(32x16 bf16) + C(16x16 f32)
// ---------------------------------------------------------------------------
static __device__ __forceinline__ v8f wmma_bf16(v16bf a, v16bf b, v8f c) {
  return __builtin_amdgcn_wmma_f32_16x16x32_bf16(
      /*neg_a=*/false, a, /*neg_b=*/false, b,
      /*c_mod=*/(short)0, c, /*reuse_a=*/false, /*reuse_b=*/false);
}

// Per-lane fragment loader for the 16-bit 16x32 A layout (B layout is the
// mirror with N in place of M). rowp = row base + k-chunk base. khalf is
// (lane>>4)*8. Lane holds K = khalf+0..7 and 16+khalf+0..7 -> two b128 loads.
static __device__ __forceinline__ v16bf load_frag(const bf16* rowp, int khalf) {
  v8bf lo = *(const v8bf*)(rowp + khalf);
  v8bf hi = *(const v8bf*)(rowp + khalf + 16);
  v16bf f;
#pragma unroll
  for (int i = 0; i < 8; ++i) { f[i] = lo[i]; f[i + 8] = hi[i]; }
  return f;
}

static __device__ __forceinline__ v8f v8f_zero() {
  v8f z = {0.f, 0.f, 0.f, 0.f, 0.f, 0.f, 0.f, 0.f};
  return z;
}

// ---------------------------------------------------------------------------
// Stage 0: f32 -> bf16 conversion (n divisible by 4)
// ---------------------------------------------------------------------------
__global__ void cvt_f32_to_bf16(const float* __restrict__ s, bf16* __restrict__ d, int n) {
  int i = (blockIdx.x * blockDim.x + threadIdx.x) * 4;
  if (i < n) {
    float4 v = *(const float4*)(s + i);
    v4bf o;
    o[0] = (bf16)v.x; o[1] = (bf16)v.y; o[2] = (bf16)v.z; o[3] = (bf16)v.w;
    *(v4bf*)(d + i) = o;
  }
}

// ---------------------------------------------------------------------------
// Stage 1: QKV projection.  y = x @ W^T  (NT GEMM, all bf16, f32 accum)
//   z==0: K -> kh [B,H,T,D]     (scatter epilogue)
//   z==1: Q -> qh [B,H,T,D]     (scatter epilogue)
//   z==2: V -> vt [B,H,D,T]     (transposed: contiguous b128 epilogue)
// Block = 8 waves (2 M x 4 N), wave tile 32x64, block tile 64x256.
// ---------------------------------------------------------------------------
__global__ __launch_bounds__(256) void qkv_gemm(
    const bf16* __restrict__ xb,
    const bf16* __restrict__ wk, const bf16* __restrict__ wq, const bf16* __restrict__ wv,
    bf16* __restrict__ kh, bf16* __restrict__ qh, bf16* __restrict__ vt) {
  const int z = blockIdx.z;
  const bf16* W = (z == 0) ? wk : (z == 1) ? wq : wv;

  const int wave = threadIdx.x >> 5, lane = threadIdx.x & 31;
  const int lrow = lane & 15, khalf = (lane >> 4) << 3;
  const int m0 = blockIdx.y * 64 + (wave >> 2) * 32;
  const int n0 = blockIdx.x * 256 + (wave & 3) * 64;

  v8f acc[2][4];
#pragma unroll
  for (int i = 0; i < 2; ++i)
#pragma unroll
    for (int j = 0; j < 4; ++j) acc[i][j] = v8f_zero();

  const bf16* ar0 = xb + (size_t)(m0 + lrow) * C_;
  const bf16* ar1 = ar0 + (size_t)16 * C_;
  const bf16* br[4];
#pragma unroll
  for (int j = 0; j < 4; ++j) br[j] = W + (size_t)(n0 + j * 16 + lrow) * C_;

#pragma unroll 2
  for (int k = 0; k < C_; k += 32) {
    v16bf a0 = load_frag(ar0 + k, khalf);
    v16bf a1 = load_frag(ar1 + k, khalf);
#pragma unroll
    for (int j = 0; j < 4; ++j) {
      v16bf b = load_frag(br[j] + k, khalf);
      acc[0][j] = wmma_bf16(a0, b, acc[0][j]);
      acc[1][j] = wmma_bf16(a1, b, acc[1][j]);
    }
  }

  if (z < 2) {
    bf16* out = (z == 0) ? kh : qh;
#pragma unroll
    for (int i = 0; i < 2; ++i)
#pragma unroll
      for (int j = 0; j < 4; ++j) {
        const int n = n0 + j * 16 + lrow;
        const int h = n >> 6, d = n & 63;
#pragma unroll
        for (int v = 0; v < 8; ++v) {
          const int m = m0 + i * 16 + khalf + v;
          const int b = m >> 10, t = m & 1023;
          out[((size_t)(b * H_ + h) * T_ + t) * D_ + d] = (bf16)acc[i][j][v];
        }
      }
  } else {
#pragma unroll
    for (int i = 0; i < 2; ++i)
#pragma unroll
      for (int j = 0; j < 4; ++j) {
        const int n = n0 + j * 16 + lrow;
        const int h = n >> 6, d = n & 63;
        const int m = m0 + i * 16 + khalf;  // 8 consecutive t, same b
        const int b = m >> 10, t = m & 1023;
        v8bf o;
#pragma unroll
        for (int v = 0; v < 8; ++v) o[v] = (bf16)acc[i][j][v];
        *(v8bf*)(vt + ((size_t)(b * H_ + h) * D_ + d) * T_ + t) = o;
      }
  }
}

// ---------------------------------------------------------------------------
// Stage 2: flash attention with the reference's k/q swap.
//   S[i,j] = k_i . q_j * D^-1/2,  mask j > i,  softmax over j,
//   Y[i,:] = sum_j P[i,j] * v_j.
// Computed transposed: S^T = Q K^T  (softmax dim j lands in register rows),
// P^T accum layout == WMMA B-fragment layout, Y^T = V^T P^T.
// One wave per 16-key i-tile; 8 waves/block.
// ---------------------------------------------------------------------------
__global__ __launch_bounds__(256) void attn_kernel(
    const bf16* __restrict__ kh, const bf16* __restrict__ qh,
    const bf16* __restrict__ vt, bf16* __restrict__ yh) {
  const int wave = threadIdx.x >> 5, lane = threadIdx.x & 31;
  const int lrow = lane & 15, khalf = (lane >> 4) << 3;

  const int g = blockIdx.x * 8 + wave;   // 8192 tiles = B*H*T/16
  const int bh = g >> 6;
  const int i0 = (g & 63) << 4;

  const bf16* Kp = kh + (size_t)bh * T_ * D_;
  const bf16* Qp = qh + (size_t)bh * T_ * D_;
  const bf16* Vp = vt + (size_t)bh * D_ * T_;
  bf16* Yp = yh + (size_t)bh * T_ * D_;

  // K B-fragments for this i-tile (column i = lane&15, contiguous d per lane)
  const v16bf kb0 = load_frag(Kp + (size_t)(i0 + lrow) * D_ + 0,  khalf);
  const v16bf kb1 = load_frag(Kp + (size_t)(i0 + lrow) * D_ + 32, khalf);

  v8f yacc[4];
#pragma unroll
  for (int dt = 0; dt < 4; ++dt) yacc[dt] = v8f_zero();

  float mrun = -1e30f, lsum = 0.f;
  const int i_g = i0 + lrow;
  const int jend = i0 + 16;  // causal: keep j <= i

  for (int j0 = 0; j0 < jend; j0 += 32) {
    // S^T tiles: rows j (16 each), cols i (16).  K-dim = d = 64 -> 2 steps.
    v8f s0 = v8f_zero(), s1 = v8f_zero();
    {
      v16bf q00 = load_frag(Qp + (size_t)(j0 + lrow) * D_ + 0,  khalf);
      v16bf q01 = load_frag(Qp + (size_t)(j0 + lrow) * D_ + 32, khalf);
      s0 = wmma_bf16(q00, kb0, s0);
      s0 = wmma_bf16(q01, kb1, s0);
      v16bf q10 = load_frag(Qp + (size_t)(j0 + 16 + lrow) * D_ + 0,  khalf);
      v16bf q11 = load_frag(Qp + (size_t)(j0 + 16 + lrow) * D_ + 32, khalf);
      s1 = wmma_bf16(q10, kb0, s1);
      s1 = wmma_bf16(q11, kb1, s1);
    }

    // scale + causal mask (per-lane: this lane owns column i = i_g)
    float p0[8], p1[8];
    float bm = -1e30f;
#pragma unroll
    for (int v = 0; v < 8; ++v) {
      const int j = j0 + khalf + v;
      float x = s0[v] * 0.125f;            // D^-1/2 = 1/8
      if (j > i_g) x = -1e30f;
      p0[v] = x; bm = fmaxf(bm, x);
    }
#pragma unroll
    for (int v = 0; v < 8; ++v) {
      const int j = j0 + 16 + khalf + v;
      float x = s1[v] * 0.125f;
      if (j > i_g) x = -1e30f;
      p1[v] = x; bm = fmaxf(bm, x);
    }
    // lanes l and l+16 hold the same column i: combine their block maxima
    bm = fmaxf(bm, __shfl_xor(bm, 16));
    const float mnew = fmaxf(mrun, bm);
    const float corr = __expf(mrun - mnew);

    // build P^T B-fragment directly (accum layout == B-fragment layout)
    float bs = 0.f;
    v16bf pb;
#pragma unroll
    for (int v = 0; v < 8; ++v) {
      float e = __expf(p0[v] - mnew); bs += e; pb[v] = (bf16)e;
    }
#pragma unroll
    for (int v = 0; v < 8; ++v) {
      float e = __expf(p1[v] - mnew); bs += e; pb[8 + v] = (bf16)e;
    }
    lsum = lsum * corr + bs;
    mrun = mnew;

#pragma unroll
    for (int dt = 0; dt < 4; ++dt) {
#pragma unroll
      for (int v = 0; v < 8; ++v) yacc[dt][v] *= corr;
      // A = V^T rows d (contiguous j per lane), K-dim = j (32)
      v16bf av = load_frag(Vp + (size_t)(dt * 16 + lrow) * T_ + j0, khalf);
      yacc[dt] = wmma_bf16(av, pb, yacc[dt]);
    }
  }

  // lanes l / l+16 summed disjoint j subsets of the same column i
  const float ltot = lsum + __shfl_xor(lsum, 16);
  const float inv = 1.f / ltot;

#pragma unroll
  for (int dt = 0; dt < 4; ++dt) {
    v8bf o;
#pragma unroll
    for (int v = 0; v < 8; ++v) o[v] = (bf16)(yacc[dt][v] * inv);
    *(v8bf*)(Yp + (size_t)i_g * D_ + dt * 16 + khalf) = o;
  }
}

// ---------------------------------------------------------------------------
// Stage 3: output projection  out = Y @ Wp^T + bp  (f32 output)
// Y is head-major [B,H,T,D]; 32-wide K chunks never cross a head (64 % 32 == 0).
// ---------------------------------------------------------------------------
__global__ __launch_bounds__(256) void proj_gemm(
    const bf16* __restrict__ yh, const bf16* __restrict__ wp,
    const float* __restrict__ bp, float* __restrict__ out) {
  const int wave = threadIdx.x >> 5, lane = threadIdx.x & 31;
  const int lrow = lane & 15, khalf = (lane >> 4) << 3;
  const int m0 = blockIdx.y * 64 + (wave >> 2) * 32;
  const int n0 = blockIdx.x * 256 + (wave & 3) * 64;

  v8f acc[2][4];
#pragma unroll
  for (int i = 0; i < 2; ++i)
#pragma unroll
    for (int j = 0; j < 4; ++j) acc[i][j] = v8f_zero();

  const int mrow = m0 + lrow;
  const int b0 = mrow >> 10, t0 = mrow & 1023;
  const bf16* br[4];
#pragma unroll
  for (int j = 0; j < 4; ++j) br[j] = wp + (size_t)(n0 + j * 16 + lrow) * C_;

#pragma unroll 2
  for (int k = 0; k < C_; k += 32) {
    const int h = k >> 6, doff = k & 63;
    const bf16* a0p = yh + ((size_t)(b0 * H_ + h) * T_ + t0) * D_ + doff;
    v16bf a0 = load_frag(a0p, khalf);
    v16bf a1 = load_frag(a0p + 16 * D_, khalf);  // t0+16, same b / same h
#pragma unroll
    for (int j = 0; j < 4; ++j) {
      v16bf b = load_frag(br[j] + k, khalf);
      acc[0][j] = wmma_bf16(a0, b, acc[0][j]);
      acc[1][j] = wmma_bf16(a1, b, acc[1][j]);
    }
  }

#pragma unroll
  for (int i = 0; i < 2; ++i)
#pragma unroll
    for (int j = 0; j < 4; ++j) {
      const int n = n0 + j * 16 + lrow;
      const float bias = bp[n];
#pragma unroll
      for (int v = 0; v < 8; ++v) {
        const int m = m0 + i * 16 + khalf + v;
        out[(size_t)m * C_ + n] = acc[i][j][v] + bias;
      }
    }
}

// ---------------------------------------------------------------------------
// Host launch
// ---------------------------------------------------------------------------
extern "C" void kernel_launch(void* const* d_in, const int* in_sizes, int n_in,
                              void* d_out, int out_size, void* d_ws, size_t ws_size,
                              hipStream_t stream) {
  const float* x  = (const float*)d_in[0];
  const float* Wk = (const float*)d_in[1];
  const float* Wq = (const float*)d_in[2];
  const float* Wv = (const float*)d_in[3];
  const float* Wp = (const float*)d_in[4];
  const float* bp = (const float*)d_in[5];
  float* out = (float*)d_out;

  char* ws = (char*)d_ws;
  const size_t SZ_X = (size_t)M_ * C_ * sizeof(bf16);   // 16 MB
  const size_t SZ_W = (size_t)C_ * C_ * sizeof(bf16);   //  2 MB
  bf16* xb  = (bf16*)(ws);
  bf16* wkb = (bf16*)(ws + SZ_X);
  bf16* wqb = (bf16*)(ws + SZ_X + 1 * SZ_W);
  bf16* wvb = (bf16*)(ws + SZ_X + 2 * SZ_W);
  bf16* wpb = (bf16*)(ws + SZ_X + 3 * SZ_W);
  bf16* kh  = (bf16*)(ws + SZ_X + 4 * SZ_W);            // [B,H,T,D]
  bf16* qh  = (bf16*)(ws + 2 * SZ_X + 4 * SZ_W);        // [B,H,T,D]
  bf16* vt  = (bf16*)(ws + 3 * SZ_X + 4 * SZ_W);        // [B,H,D,T]
  bf16* yh  = (bf16*)(ws + 4 * SZ_X + 4 * SZ_W);        // [B,H,T,D]

  // Stage 0: conversions
  {
    const int nx = M_ * C_, nw = C_ * C_;
    cvt_f32_to_bf16<<<dim3((nx / 4 + 255) / 256), dim3(256), 0, stream>>>(x,  xb,  nx);
    cvt_f32_to_bf16<<<dim3((nw / 4 + 255) / 256), dim3(256), 0, stream>>>(Wk, wkb, nw);
    cvt_f32_to_bf16<<<dim3((nw / 4 + 255) / 256), dim3(256), 0, stream>>>(Wq, wqb, nw);
    cvt_f32_to_bf16<<<dim3((nw / 4 + 255) / 256), dim3(256), 0, stream>>>(Wv, wvb, nw);
    cvt_f32_to_bf16<<<dim3((nw / 4 + 255) / 256), dim3(256), 0, stream>>>(Wp, wpb, nw);
  }

  // Stage 1: QKV projection (z selects K/Q/V)
  qkv_gemm<<<dim3(C_ / 256, M_ / 64, 3), dim3(256), 0, stream>>>(
      xb, wkb, wqb, wvb, kh, qh, vt);

  // Stage 2: causal "attention" (with the reference's k/q role swap)
  attn_kernel<<<dim3((B_ * H_ * (T_ / 16)) / 8), dim3(256), 0, stream>>>(
      kh, qh, vt, yh);

  // Stage 3: output projection + bias (f32 out)
  proj_gemm<<<dim3(C_ / 256, M_ / 64), dim3(256), 0, stream>>>(yh, wpb, bp, out);
}
